// GIN_71665824301262
// MI455X (gfx1250) — compile-verified, matrix operand order checked by reading, not run on hardware
//
#include <hip/hip_runtime.h>
#include <hip/hip_bf16.h>
#include <math.h>

typedef _Float16 half4v __attribute__((ext_vector_type(4)));
typedef _Float16 v8h    __attribute__((ext_vector_type(8)));
typedef _Float16 v16h   __attribute__((ext_vector_type(16)));
typedef float    v8f    __attribute__((ext_vector_type(8)));

#define D_DIM 384
#define H_DIM 768

__device__ __forceinline__ float gelu_exact(float v) {
    return 0.5f * v * (1.0f + erff(v * 0.70710678118654752f));
}

// ---------------------------------------------------------------------------
// BatchNorm column statistics: thread = column (coalesced), block = row chunk
// ---------------------------------------------------------------------------
__global__ void k_colstats(const float* __restrict__ x, int N,
                           float* __restrict__ sum, float* __restrict__ sumsq) {
    int col = threadIdx.x;
    int chunk = (N + gridDim.x - 1) / gridDim.x;
    int r0 = blockIdx.x * chunk;
    int r1 = min(N, r0 + chunk);
    float s = 0.f, s2 = 0.f;
    for (int r = r0; r < r1; ++r) {
        float v = x[(size_t)r * D_DIM + col];
        s += v;
        s2 += v * v;
    }
    atomicAdd(&sum[col], s);
    atomicAdd(&sumsq[col], s2);
}

__global__ void k_bn_finalize(const float* __restrict__ sum, const float* __restrict__ sumsq,
                              const float* __restrict__ gamma, const float* __restrict__ beta,
                              int N, float* __restrict__ scale, float* __restrict__ shift) {
    int c = threadIdx.x;
    float inv = 1.0f / (float)N;
    float mu  = sum[c] * inv;
    float var = sumsq[c] * inv - mu * mu;
    float rs  = rsqrtf(var + 1e-5f);
    float sc  = rs * gamma[c];
    scale[c] = sc;
    shift[c] = beta[c] - mu * sc;
}

__global__ void k_bn_apply(const float* __restrict__ x, const float* __restrict__ scale,
                           const float* __restrict__ shift, size_t n4, float* __restrict__ h) {
    size_t i = blockIdx.x * (size_t)blockDim.x + threadIdx.x;
    size_t stride = (size_t)gridDim.x * blockDim.x;
    const float4* xv = (const float4*)x;
    float4* hv = (float4*)h;
    for (; i < n4; i += stride) {
        int c4 = (int)(i % (D_DIM / 4)) * 4;
        float4 v = xv[i];
        float4 o;
        o.x = v.x * scale[c4 + 0] + shift[c4 + 0];
        o.y = v.y * scale[c4 + 1] + shift[c4 + 1];
        o.z = v.z * scale[c4 + 2] + shift[c4 + 2];
        o.w = v.w * scale[c4 + 3] + shift[c4 + 3];
        hv[i] = o;
    }
}

__global__ void k_make_z16(const float* __restrict__ h, const float* __restrict__ agg,
                           const float* __restrict__ eps_p, size_t n4,
                           _Float16* __restrict__ z16) {
    float c1 = 1.0f + eps_p[0];
    size_t i = blockIdx.x * (size_t)blockDim.x + threadIdx.x;
    size_t stride = (size_t)gridDim.x * blockDim.x;
    const float4* hv = (const float4*)h;
    const float4* av = (const float4*)agg;
    half4v* zv = (half4v*)z16;
    for (; i < n4; i += stride) {
        float4 a = hv[i];
        float4 b = av[i];
        half4v o = { (_Float16)(c1 * a.x + b.x), (_Float16)(c1 * a.y + b.y),
                     (_Float16)(c1 * a.z + b.z), (_Float16)(c1 * a.w + b.w) };
        zv[i] = o;
    }
}

// ---------------------------------------------------------------------------
// WMMA GEMM: 128x64 block tile, 8 waves, each wave 32x32 (2x2 wmma 16x16x32).
// A-tile edge rows are address-CLAMPED (not zero-filled): garbage only lands
// in output rows that the epilogue masks with (gr < M), so the K-loop is
// completely branch-free.
// MODE 0: A=f32 edge_attr, epi = +b_edge + h[src], relu, atomicAdd(agg[dst])
// MODE 1: A=f16 z16,       epi = +b1, gelu, store f16 g16
// MODE 2: A=f16 g16,       epi = +b2, out = x + gelu(.)  (f32)
// ---------------------------------------------------------------------------
template <int MODE>
__global__ __launch_bounds__(256) void k_gemm(
    const void* __restrict__ Aptr, const float* __restrict__ Bmat,
    const float* __restrict__ bias, int M, int K, int Nn,
    const float* __restrict__ hmat, const int* __restrict__ srcIdx,
    const int* __restrict__ dstIdx, float* __restrict__ agg,
    _Float16* __restrict__ Gout, const float* __restrict__ xres,
    float* __restrict__ fout) {
    __shared__ _Float16 As[128 * 32];   // [m][k] row-major
    __shared__ _Float16 Bs[64 * 32];    // transposed: [n][k]

    const int tid  = threadIdx.x;
    const int lane = tid & 31;
    const int wave = tid >> 5;
    const int wm   = wave >> 1;         // 0..3  (M sub-block of 32)
    const int wn   = wave & 1;          // 0..1  (N sub-block of 32)
    const int rowBase = blockIdx.x * 128;
    const int n0      = blockIdx.y * 64;
    const int lh = lane & 15;
    const int kb = (lane < 16) ? 0 : 8; // K-half select per ISA A/B layout

    v8f acc[2][2];
    #pragma unroll
    for (int i = 0; i < 2; ++i)
        #pragma unroll
        for (int j = 0; j < 2; ++j)
            acc[i][j] = (v8f){};

    // per-thread tile-loader coordinates (constant across K loop)
    const int arow = tid >> 3;          // A: thread covers rows tid/8 + it*32
    const int akq  = tid & 7;           //    float4/half4 column within 32-wide K
    const int bk   = tid >> 4;          // B: k row  (0..15, +16 on second it)
    const int bnq  = tid & 15;          //    float4 column within 64-wide N

    // clamped A row addresses (branch-free edge handling)
    int agr[4];
    #pragma unroll
    for (int it = 0; it < 4; ++it)
        agr[it] = min(rowBase + arow + it * 32, M - 1);

    for (int k0 = 0; k0 < K; k0 += 32) {
        __syncthreads();

        // ---- phase 1: issue all global loads into registers (pipelined) ----
        float4 bv0 = *(const float4*)(Bmat + (size_t)(k0 + bk) * Nn + n0 + bnq * 4);
        float4 bv1 = *(const float4*)(Bmat + (size_t)(k0 + bk + 16) * Nn + n0 + bnq * 4);

        if (MODE == 0) {
            const float* Af = (const float*)Aptr;
            float4 av[4];
            #pragma unroll
            for (int it = 0; it < 4; ++it)
                av[it] = *(const float4*)(Af + (size_t)agr[it] * K + k0 + akq * 4);
            #pragma unroll
            for (int it = 0; it < 4; ++it) {
                half4v hv = { (_Float16)av[it].x, (_Float16)av[it].y,
                              (_Float16)av[it].z, (_Float16)av[it].w };
                *(half4v*)(&As[(arow + it * 32) * 32 + akq * 4]) = hv;
            }
        } else {
            const _Float16* Ah = (const _Float16*)Aptr;
            half4v av[4];
            #pragma unroll
            for (int it = 0; it < 4; ++it)
                av[it] = *(const half4v*)(Ah + (size_t)agr[it] * K + k0 + akq * 4);
            #pragma unroll
            for (int it = 0; it < 4; ++it)
                *(half4v*)(&As[(arow + it * 32) * 32 + akq * 4]) = av[it];
        }

        // ---- phase 2: B tile -> LDS transposed [n][k] ----
        Bs[(bnq * 4 + 0) * 32 + bk] = (_Float16)bv0.x;
        Bs[(bnq * 4 + 1) * 32 + bk] = (_Float16)bv0.y;
        Bs[(bnq * 4 + 2) * 32 + bk] = (_Float16)bv0.z;
        Bs[(bnq * 4 + 3) * 32 + bk] = (_Float16)bv0.w;
        Bs[(bnq * 4 + 0) * 32 + bk + 16] = (_Float16)bv1.x;
        Bs[(bnq * 4 + 1) * 32 + bk + 16] = (_Float16)bv1.y;
        Bs[(bnq * 4 + 2) * 32 + bk + 16] = (_Float16)bv1.z;
        Bs[(bnq * 4 + 3) * 32 + bk + 16] = (_Float16)bv1.w;
        __syncthreads();

        // ---- phase 3: fragments (two contiguous 16B LDS loads each) + WMMA ----
        v16h af[2], bf[2];
        #pragma unroll
        for (int i = 0; i < 2; ++i) {
            int m = wm * 32 + i * 16 + lh;
            v8h lo = *(const v8h*)(&As[m * 32 + kb]);
            v8h hi = *(const v8h*)(&As[m * 32 + 16 + kb]);
            af[i] = __builtin_shufflevector(lo, hi, 0,1,2,3,4,5,6,7,8,9,10,11,12,13,14,15);
        }
        #pragma unroll
        for (int j = 0; j < 2; ++j) {
            int n = wn * 32 + j * 16 + lh;
            v8h lo = *(const v8h*)(&Bs[n * 32 + kb]);
            v8h hi = *(const v8h*)(&Bs[n * 32 + 16 + kb]);
            bf[j] = __builtin_shufflevector(lo, hi, 0,1,2,3,4,5,6,7,8,9,10,11,12,13,14,15);
        }
        #pragma unroll
        for (int i = 0; i < 2; ++i)
            #pragma unroll
            for (int j = 0; j < 2; ++j)
                acc[i][j] = __builtin_amdgcn_wmma_f32_16x16x32_f16(
                    false, af[i], false, bf[j], (short)0, acc[i][j], false, false);
    }

    // ---- epilogue: C/D layout lane l -> col = l&15, rows = vr + (l<16?0:8)
    const int rsel = (lane < 16) ? 0 : 8;
    float bvj[2];
    #pragma unroll
    for (int j = 0; j < 2; ++j)
        bvj[j] = bias[n0 + wn * 32 + j * 16 + lh];

    #pragma unroll
    for (int i = 0; i < 2; ++i) {
        #pragma unroll
        for (int vr = 0; vr < 8; ++vr) {
            int gr = rowBase + wm * 32 + i * 16 + vr + rsel;
            if (gr < M) {
                if (MODE == 0) {
                    int s = srcIdx[gr];
                    int d = dstIdx[gr];
                    #pragma unroll
                    for (int j = 0; j < 2; ++j) {
                        int gc = n0 + wn * 32 + j * 16 + lh;
                        float val = acc[i][j][vr] + bvj[j];
                        val += hmat[(size_t)s * D_DIM + gc];
                        val = fmaxf(val, 0.0f);
                        atomicAdd(&agg[(size_t)d * D_DIM + gc], val);
                    }
                } else if (MODE == 1) {
                    #pragma unroll
                    for (int j = 0; j < 2; ++j) {
                        int gc = n0 + wn * 32 + j * 16 + lh;
                        float val = acc[i][j][vr] + bvj[j];
                        Gout[(size_t)gr * Nn + gc] = (_Float16)gelu_exact(val);
                    }
                } else {
                    #pragma unroll
                    for (int j = 0; j < 2; ++j) {
                        int gc = n0 + wn * 32 + j * 16 + lh;
                        float val = acc[i][j][vr] + bvj[j];
                        size_t o = (size_t)gr * (size_t)Nn + gc;
                        fout[o] = xres[o] + gelu_exact(val);
                    }
                }
            }
        }
    }
}

// ---------------------------------------------------------------------------
extern "C" void kernel_launch(void* const* d_in, const int* in_sizes, int n_in,
                              void* d_out, int out_size, void* d_ws, size_t ws_size,
                              hipStream_t stream) {
    const float* x      = (const float*)d_in[0];
    const float* ea     = (const float*)d_in[1];
    const float* gamma  = (const float*)d_in[2];
    const float* beta   = (const float*)d_in[3];
    const float* eps    = (const float*)d_in[4];
    const float* W_edge = (const float*)d_in[5];
    const float* b_edge = (const float*)d_in[6];
    const float* W1     = (const float*)d_in[7];
    const float* b1     = (const float*)d_in[8];
    const float* W2     = (const float*)d_in[9];
    const float* b2     = (const float*)d_in[10];
    const int*   ei     = (const int*)d_in[11];

    const int N = in_sizes[0] / D_DIM;   // 100000
    const int E = in_sizes[1] / D_DIM;   // 200000

    // workspace layout
    char* w = (char*)d_ws;
    float* sum   = (float*)w;            // 384
    float* sumsq = sum + D_DIM;          // 384
    float* scale = sumsq + D_DIM;        // 384
    float* shift = scale + D_DIM;        // 384
    float*    h   = (float*)(w + 8192);                  // N*D f32
    float*    agg = h + (size_t)N * D_DIM;               // N*D f32
    _Float16* z16 = (_Float16*)(agg + (size_t)N * D_DIM);// N*D f16
    _Float16* g16 = z16 + (size_t)N * D_DIM;             // N*H f16

    hipMemsetAsync(sum, 0, 2 * D_DIM * sizeof(float), stream);
    hipMemsetAsync(agg, 0, (size_t)N * D_DIM * sizeof(float), stream);

    k_colstats<<<256, D_DIM, 0, stream>>>(x, N, sum, sumsq);
    k_bn_finalize<<<1, D_DIM, 0, stream>>>(sum, sumsq, gamma, beta, N, scale, shift);

    size_t n4 = (size_t)N * D_DIM / 4;
    k_bn_apply<<<2048, 256, 0, stream>>>(x, scale, shift, n4, h);

    // fused edge GEMM + message + scatter-add (agg lives in L2)
    dim3 g0((unsigned)((E + 127) / 128), D_DIM / 64);
    k_gemm<0><<<g0, 256, 0, stream>>>(ea, W_edge, b_edge, E, D_DIM, D_DIM,
                                      h, ei, ei + E, agg, nullptr, nullptr, nullptr);

    k_make_z16<<<2048, 256, 0, stream>>>(h, agg, eps, n4, z16);

    dim3 g1((unsigned)((N + 127) / 128), H_DIM / 64);
    k_gemm<1><<<g1, 256, 0, stream>>>(z16, W1, b1, N, D_DIM, H_DIM,
                                      nullptr, nullptr, nullptr, nullptr, g16, nullptr, nullptr);

    dim3 g2((unsigned)((N + 127) / 128), D_DIM / 64);
    k_gemm<2><<<g2, 256, 0, stream>>>(g16, W2, b2, N, H_DIM, D_DIM,
                                      nullptr, nullptr, nullptr, nullptr, nullptr, x, (float*)d_out);
}
